// NeighborhoodTransformer_24361054502955
// MI455X (gfx1250) — matrix operand order repaired and checked by reading, not compiled, not run on hardware
//
#include <hip/hip_runtime.h>

// ---------------------------------------------------------------------------
// NeighborhoodTransformer on gfx1250 (MI455X): one node per workgroup,
// whole K=32 sequence resident in LDS, all GEMMs on v_wmma_f32_16x16x32_f16.
// ---------------------------------------------------------------------------

#define KNB     32
#define DMODEL  256
#define NHEADS  8
#define DHEAD   32
#define NLAYERS 3
#define QKVDIM  768

// LDS row strides (halves / floats), padded to keep 16B alignment per chunk
#define SX_STRIDE   264   // f16 [32][264]
#define SQKV_STRIDE 776   // f16 [32][776]
#define SS_STRIDE   33    // f32 [8][32][33]
#define SP_STRIDE   40    // f16 [8][32][40]
#define SVT_STRIDE  40    // f16 [256][40]

#define SX_ELEMS   (32 * SX_STRIDE)
#define SQKV_ELEMS (32 * SQKV_STRIDE)
#define SS_ELEMS   (NHEADS * 32 * SS_STRIDE)
#define SP_ELEMS   (NHEADS * 32 * SP_STRIDE)
#define SVT_ELEMS  (256 * SVT_STRIDE)
#define SA_ELEMS   (32 * SX_STRIDE)
#define ST_ELEMS   (32 * SX_STRIDE)

static constexpr size_t LDS_BYTES =
    2ull * (SX_ELEMS + SQKV_ELEMS + SP_ELEMS + SVT_ELEMS + SA_ELEMS + ST_ELEMS) +
    4ull * SS_ELEMS;   // = 175,104 B  (< 320 KB/WGP)

typedef __attribute__((ext_vector_type(8)))  _Float16 v8h;
typedef __attribute__((ext_vector_type(16))) _Float16 v16h;
typedef __attribute__((ext_vector_type(8)))  float    v8f;

__device__ __forceinline__ v8f wmma16(v16h a, v16h b, v8f c) {
  // D(16x16 f32) = A(16x32 f16) * B(32x16 f16) + C
  return __builtin_amdgcn_wmma_f32_16x16x32_f16(false, a, false, b, (short)0, c,
                                                false, false);
}

// A fragment (16x32 f16): lane holds row (row0 + lane&15); its 16 halves map to
// K = {8g..8g+7, 16+8g..16+8g+7}, g = lane>>4.  Two 16B chunks.
__device__ __forceinline__ v16h fragA(const _Float16* base, int stride,
                                      int row0, int col0, int lane) {
  const int r = row0 + (lane & 15);
  const int g = lane >> 4;
  const _Float16* p = base + (size_t)r * stride + col0 + 8 * g;
  v8h lo = *(const v8h*)(p);
  v8h hi = *(const v8h*)(p + 16);
  v16h o;
#pragma unroll
  for (int i = 0; i < 8; ++i) { o[i] = lo[i]; o[i + 8] = hi[i]; }
  return o;
}

// B fragment (32x16 f16): lane holds column (n0 + lane&15); its 16 halves map
// to K = {8g..8g+7, 16+8g..16+8g+7}.  base is row-major [n][k] (i.e. W[out][in]).
__device__ __forceinline__ v16h fragB(const _Float16* base, int stride,
                                      int n0, int k0, int lane) {
  const int nn = n0 + (lane & 15);
  const int g = lane >> 4;
  const _Float16* p = base + (size_t)nn * stride + k0 + 8 * g;
  v8h lo = *(const v8h*)(p);
  v8h hi = *(const v8h*)(p + 16);
  v16h o;
#pragma unroll
  for (int i = 0; i < 8; ++i) { o[i] = lo[i]; o[i + 8] = hi[i]; }
  return o;
}

__global__ void cvt_f32_to_f16_kernel(const float* __restrict__ src,
                                      _Float16* __restrict__ dst, int n) {
  int i = blockIdx.x * blockDim.x + threadIdx.x;
  if (i < n) dst[i] = (_Float16)src[i];
}

__global__ __launch_bounds__(256, 1)
void nbr_transformer_kernel(
    const float* __restrict__ hfeat, const float* __restrict__ efeat,
    const int* __restrict__ src_idx,
    const _Float16* __restrict__ w16_qkv,   // [L][768][256]
    const _Float16* __restrict__ w16_out,   // [L][256][256]
    const _Float16* __restrict__ w16_ff1,   // [L][256][256]
    const _Float16* __restrict__ w16_ff2,   // [L][256][256]
    const float* __restrict__ b_qkv, const float* __restrict__ b_out,
    const float* __restrict__ b_ff1, const float* __restrict__ b_ff2,
    const float* __restrict__ ln1_g, const float* __restrict__ ln1_b,
    const float* __restrict__ ln2_g, const float* __restrict__ ln2_b,
    float* __restrict__ out) {
  extern __shared__ char smem[];
  _Float16* sX   = (_Float16*)smem;            // current sequence [32][264]
  _Float16* sQKV = sX + SX_ELEMS;              // packed q|k|v [32][776]
  float*    sS   = (float*)(sQKV + SQKV_ELEMS);// per-head scores
  _Float16* sP   = (_Float16*)(sS + SS_ELEMS); // per-head softmax probs
  _Float16* sVT  = sP + SP_ELEMS;              // V transposed [256][40]
  _Float16* sA   = sVT + SVT_ELEMS;            // attn-out / ff2-out [32][264]
  _Float16* sT   = sA + SA_ELEMS;              // residual / relu buf [32][264]

  const int node = blockIdx.x;
  const int tid  = threadIdx.x;
  const int lane = tid & 31;
  const int wave = tid >> 5;
  const int l15  = lane & 15;
  const int g    = lane >> 4;

  // ---- gather: msgs[k][d] = h[src[k]][d] + e[node][k][d]  -> f16 in sX ----
  {
    const int* si = src_idx + node * KNB;
    for (int idx = tid; idx < KNB * (DMODEL / 4); idx += 256) {
      const int k  = idx >> 6;           // 64 float4 per row
      const int d4 = (idx & 63) * 4;
      const int s  = si[k];
      const float4 hv = *(const float4*)(hfeat + (size_t)s * DMODEL + d4);
      const float4 ev =
          *(const float4*)(efeat + ((size_t)node * KNB + k) * DMODEL + d4);
      _Float16* dst = sX + k * SX_STRIDE + d4;
      dst[0] = (_Float16)(hv.x + ev.x);
      dst[1] = (_Float16)(hv.y + ev.y);
      dst[2] = (_Float16)(hv.z + ev.z);
      dst[3] = (_Float16)(hv.w + ev.w);
    }
  }
  __syncthreads();

  for (int l = 0; l < NLAYERS; ++l) {
    const _Float16* Wqkv = w16_qkv + (size_t)l * QKVDIM * DMODEL;
    const _Float16* Wo   = w16_out + (size_t)l * DMODEL * DMODEL;
    const _Float16* W1   = w16_ff1 + (size_t)l * DMODEL * DMODEL;
    const _Float16* W2   = w16_ff2 + (size_t)l * DMODEL * DMODEL;
    const float* Bqkv = b_qkv + l * QKVDIM;
    const float* Bo   = b_out + l * DMODEL;
    const float* B1   = b_ff1 + l * DMODEL;
    const float* B2   = b_ff2 + l * DMODEL;

    // ---- QKV projection: sQKV = sX @ Wqkv^T + b  (96 tiles, 12 per wave) ----
#pragma unroll 1
    for (int rep = 0; rep < 12; ++rep) {
      const int t  = wave + 8 * rep;
      const int tm = (t & 1) * 16;
      const int tn = (t >> 1) * 16;
      const float bias = Bqkv[tn + l15];
      v8f acc;
#pragma unroll
      for (int i = 0; i < 8; ++i) acc[i] = bias;
#pragma unroll
      for (int kb = 0; kb < 8; ++kb) {
        v16h a = fragA(sX, SX_STRIDE, tm, kb * 32, lane);
        v16h b = fragB(Wqkv, DMODEL, tn, kb * 32, lane);
        acc = wmma16(a, b, acc);
      }
      const int col = tn + l15;
#pragma unroll
      for (int i = 0; i < 8; ++i)
        sQKV[(tm + i + 8 * g) * SQKV_STRIDE + col] = (_Float16)acc[i];
      if (col >= 2 * DMODEL) {  // V range: also keep a transposed copy
        _Float16* dst = sVT + (col - 2 * DMODEL) * SVT_STRIDE + tm + 8 * g;
#pragma unroll
        for (int i = 0; i < 8; ++i) dst[i] = (_Float16)acc[i];
      }
    }
    __syncthreads();

    // ---- attention: one head per wave (wave-private LDS, no block barrier) --
    {
      const int qo = wave * DHEAD;           // Q column offset in sQKV
      const int ko = DMODEL + wave * DHEAD;  // K column offset in sQKV
      float*    Sh = sS + wave * (32 * SS_STRIDE);
      _Float16* Ph = sP + wave * (32 * SP_STRIDE);

      // scores = (Q @ K^T) * scale ; B[d][n] = Kmat[n][d] contiguous in sQKV rows
#pragma unroll
      for (int tmi = 0; tmi < 2; ++tmi)
#pragma unroll
        for (int tni = 0; tni < 2; ++tni) {
          v8f acc;
#pragma unroll
          for (int i = 0; i < 8; ++i) acc[i] = 0.0f;
          v16h a = fragA(sQKV, SQKV_STRIDE, tmi * 16, qo, lane);
          v16h b = fragB(sQKV, SQKV_STRIDE, tni * 16, ko, lane);
          acc = wmma16(a, b, acc);
          const int col = tni * 16 + l15;
#pragma unroll
          for (int i = 0; i < 8; ++i)
            Sh[(tmi * 16 + i + 8 * g) * SS_STRIDE + col] =
                acc[i] * 0.17677669529663687f;  // 1/sqrt(32)
        }
      __builtin_amdgcn_wave_barrier();

      // row softmax: lane r owns row r
      {
        float* Sr = Sh + lane * SS_STRIDE;
        float mx = -3.0e38f;
#pragma unroll
        for (int c = 0; c < 32; ++c) mx = fmaxf(mx, Sr[c]);
        float sum = 0.0f;
#pragma unroll
        for (int c = 0; c < 32; ++c) {
          float ev = __expf(Sr[c] - mx);
          Sr[c] = ev;
          sum += ev;
        }
        const float inv = 1.0f / sum;
        _Float16* Pr = Ph + lane * SP_STRIDE;
#pragma unroll
        for (int c = 0; c < 32; ++c) Pr[c] = (_Float16)(Sr[c] * inv);
      }
      __builtin_amdgcn_wave_barrier();

      // attn_out = P @ V  (V^T resident in sVT so B is contiguous)
#pragma unroll
      for (int tmi = 0; tmi < 2; ++tmi)
#pragma unroll
        for (int tni = 0; tni < 2; ++tni) {
          v8f acc;
#pragma unroll
          for (int i = 0; i < 8; ++i) acc[i] = 0.0f;
          v16h a = fragA(Ph, SP_STRIDE, tmi * 16, 0, lane);
          v16h b = fragB(sVT, SVT_STRIDE, wave * DHEAD + tni * 16, 0, lane);
          acc = wmma16(a, b, acc);
          const int col = wave * DHEAD + tni * 16 + l15;
#pragma unroll
          for (int i = 0; i < 8; ++i)
            sA[(tmi * 16 + i + 8 * g) * SX_STRIDE + col] = (_Float16)acc[i];
        }
    }
    __syncthreads();

    // ---- out projection + residual -> sT ----
#pragma unroll 1
    for (int rep = 0; rep < 4; ++rep) {
      const int t  = wave + 8 * rep;
      const int tm = (t & 1) * 16;
      const int tn = (t >> 1) * 16;
      const float bias = Bo[tn + l15];
      v8f acc;
#pragma unroll
      for (int i = 0; i < 8; ++i) acc[i] = bias;
#pragma unroll
      for (int kb = 0; kb < 8; ++kb) {
        v16h a = fragA(sA, SX_STRIDE, tm, kb * 32, lane);
        v16h b = fragB(Wo, DMODEL, tn, kb * 32, lane);
        acc = wmma16(a, b, acc);
      }
      const int col = tn + l15;
#pragma unroll
      for (int i = 0; i < 8; ++i) {
        const int m = tm + i + 8 * g;
        const float r = (float)sX[m * SX_STRIDE + col];
        sT[m * SX_STRIDE + col] = (_Float16)(r + acc[i]);
      }
    }
    __syncthreads();

    // ---- LayerNorm 1: sT -> sX ----
    if (tid < 32) {
      const _Float16* row = sT + tid * SX_STRIDE;
      float mu = 0.0f;
      for (int d = 0; d < DMODEL; ++d) mu += (float)row[d];
      mu *= (1.0f / DMODEL);
      float var = 0.0f;
      for (int d = 0; d < DMODEL; ++d) {
        float dv = (float)row[d] - mu;
        var += dv * dv;
      }
      var *= (1.0f / DMODEL);
      const float inv = rsqrtf(var + 1e-5f);
      const float* gg = ln1_g + l * DMODEL;
      const float* bb = ln1_b + l * DMODEL;
      _Float16* orow = sX + tid * SX_STRIDE;
      for (int d = 0; d < DMODEL; ++d)
        orow[d] = (_Float16)(((float)row[d] - mu) * inv * gg[d] + bb[d]);
    }
    __syncthreads();

    // ---- FF1 + ReLU : sX -> sT ----
#pragma unroll 1
    for (int rep = 0; rep < 4; ++rep) {
      const int t  = wave + 8 * rep;
      const int tm = (t & 1) * 16;
      const int tn = (t >> 1) * 16;
      const float bias = B1[tn + l15];
      v8f acc;
#pragma unroll
      for (int i = 0; i < 8; ++i) acc[i] = bias;
#pragma unroll
      for (int kb = 0; kb < 8; ++kb) {
        v16h a = fragA(sX, SX_STRIDE, tm, kb * 32, lane);
        v16h b = fragB(W1, DMODEL, tn, kb * 32, lane);
        acc = wmma16(a, b, acc);
      }
      const int col = tn + l15;
#pragma unroll
      for (int i = 0; i < 8; ++i)
        sT[(tm + i + 8 * g) * SX_STRIDE + col] = (_Float16)fmaxf(acc[i], 0.0f);
    }
    __syncthreads();

    // ---- FF2 + residual : sT -> sA ----
#pragma unroll 1
    for (int rep = 0; rep < 4; ++rep) {
      const int t  = wave + 8 * rep;
      const int tm = (t & 1) * 16;
      const int tn = (t >> 1) * 16;
      const float bias = B2[tn + l15];
      v8f acc;
#pragma unroll
      for (int i = 0; i < 8; ++i) acc[i] = bias;
#pragma unroll
      for (int kb = 0; kb < 8; ++kb) {
        v16h a = fragA(sT, SX_STRIDE, tm, kb * 32, lane);
        v16h b = fragB(W2, DMODEL, tn, kb * 32, lane);
        acc = wmma16(a, b, acc);
      }
      const int col = tn + l15;
#pragma unroll
      for (int i = 0; i < 8; ++i) {
        const int m = tm + i + 8 * g;
        const float r = (float)sX[m * SX_STRIDE + col];
        sA[m * SX_STRIDE + col] = (_Float16)(r + acc[i]);
      }
    }
    __syncthreads();

    // ---- LayerNorm 2: sA -> sX ----
    if (tid < 32) {
      const _Float16* row = sA + tid * SX_STRIDE;
      float mu = 0.0f;
      for (int d = 0; d < DMODEL; ++d) mu += (float)row[d];
      mu *= (1.0f / DMODEL);
      float var = 0.0f;
      for (int d = 0; d < DMODEL; ++d) {
        float dv = (float)row[d] - mu;
        var += dv * dv;
      }
      var *= (1.0f / DMODEL);
      const float inv = rsqrtf(var + 1e-5f);
      const float* gg = ln2_g + l * DMODEL;
      const float* bb = ln2_b + l * DMODEL;
      _Float16* orow = sX + tid * SX_STRIDE;
      for (int d = 0; d < DMODEL; ++d)
        orow[d] = (_Float16)(((float)row[d] - mu) * inv * gg[d] + bb[d]);
    }
    __syncthreads();
  }

  // readout: transformer output at sequence position 0
  out[(size_t)node * DMODEL + tid] = (float)sX[tid];
}

extern "C" void kernel_launch(void* const* d_in, const int* in_sizes, int n_in,
                              void* d_out, int out_size, void* d_ws,
                              size_t ws_size, hipStream_t stream) {
  const float* hfeat = (const float*)d_in[0];
  const float* efeat = (const float*)d_in[1];
  const int*   sidx  = (const int*)d_in[2];
  const float* wqkv  = (const float*)d_in[3];
  const float* bqkv  = (const float*)d_in[4];
  const float* wo    = (const float*)d_in[5];
  const float* bo    = (const float*)d_in[6];
  const float* w1    = (const float*)d_in[7];
  const float* b1    = (const float*)d_in[8];
  const float* w2    = (const float*)d_in[9];
  const float* b2    = (const float*)d_in[10];
  const float* ln1g  = (const float*)d_in[11];
  const float* ln1b  = (const float*)d_in[12];
  const float* ln2g  = (const float*)d_in[13];
  const float* ln2b  = (const float*)d_in[14];
  float* out = (float*)d_out;

  const int n_nodes = in_sizes[0] / DMODEL;

  // f16 weight copies in workspace
  const int nqkv = NLAYERS * QKVDIM * DMODEL;   // 589,824
  const int nsq  = NLAYERS * DMODEL * DMODEL;   // 196,608
  _Float16* w16qkv = (_Float16*)d_ws;
  _Float16* w16o   = w16qkv + nqkv;
  _Float16* w16f1  = w16o + nsq;
  _Float16* w16f2  = w16f1 + nsq;

  cvt_f32_to_f16_kernel<<<(nqkv + 255) / 256, 256, 0, stream>>>(wqkv, w16qkv, nqkv);
  cvt_f32_to_f16_kernel<<<(nsq + 255) / 256, 256, 0, stream>>>(wo, w16o, nsq);
  cvt_f32_to_f16_kernel<<<(nsq + 255) / 256, 256, 0, stream>>>(w1, w16f1, nsq);
  cvt_f32_to_f16_kernel<<<(nsq + 255) / 256, 256, 0, stream>>>(w2, w16f2, nsq);

  hipFuncSetAttribute((const void*)nbr_transformer_kernel,
                      hipFuncAttributeMaxDynamicSharedMemorySize,
                      (int)LDS_BYTES);

  nbr_transformer_kernel<<<n_nodes, 256, LDS_BYTES, stream>>>(
      hfeat, efeat, sidx, w16qkv, w16o, w16f1, w16f2, bqkv, bo, b1, b2, ln1g,
      ln1b, ln2g, ln2b, out);
}
